// FlippedQuanv3x3_20521353740762
// MI455X (gfx1250) — matrix operand order repaired and checked by reading, not compiled, not run on hardware
//
#include <hip/hip_runtime.h>
#include <hip/hip_bf16.h>

typedef __attribute__((ext_vector_type(16))) _Float16 v16h;
typedef __attribute__((ext_vector_type(8)))  _Float16 v8h;
typedef __attribute__((ext_vector_type(8)))  float    v8f;

#define KP   160        // K = 16 ch * 9 taps = 144, padded to 160 (5 x 32)
#define CIN  16
#define HW   128
#define COUT 64

// Dynamic-LDS layout (base offset 0: kernel has no static LDS)
#define LDS_W     0u         // f16 Weff   [64][160]      20480 B
#define LDS_RAW   20480u     // f32 raw    [16][3][128]   24576 B
#define LDS_B     45056u     // f16 im2col [128][160]     40960 B
#define LDS_BIAS  86016u     // f32 bias   [64]             256 B
#define LDS_TOTAL 86272u

// ---------------------------------------------------------------------------
// CDNA5 async global->LDS copy (ASYNCcnt-tracked), per-lane 16B transfer.
// ---------------------------------------------------------------------------
__device__ __forceinline__ void async_copy_b128(unsigned lds_off, const void* g) {
    asm volatile("global_load_async_to_lds_b128 %0, %1, off"
                 :: "v"(lds_off), "v"(g) : "memory");
}
__device__ __forceinline__ void wait_async0() {
    asm volatile("s_wait_asynccnt 0" ::: "memory");
}

// ---------------------------------------------------------------------------
// Setup: fold Pauli features into conv weights.
// Weff[o][c*9+j] = 0.25 * ( sum_i weight[o][c][i]*A[i][j] + t0[j] ),  fp16.
// ---------------------------------------------------------------------------
__device__ inline void pauli2(int s, float re[2][2], float im[2][2]) {
    for (int r = 0; r < 2; ++r)
        for (int c = 0; c < 2; ++c) { re[r][c] = 0.f; im[r][c] = 0.f; }
    switch (s) {
        case 0: re[0][0] = 1.f; re[1][1] = 1.f; break;   // I
        case 1: re[0][1] = 1.f; re[1][0] = 1.f; break;   // X
        case 2: im[0][1] = -1.f; im[1][0] = 1.f; break;  // Y
        case 3: re[0][0] = 1.f; re[1][1] = -1.f; break;  // Z
    }
}

__global__ void weff_setup_kernel(const float* __restrict__ weight,
                                  _Float16* __restrict__ weff) {
    __shared__ float sA[15][9];
    if (threadIdx.x == 0) {
        for (int pi = 1; pi <= 15; ++pi) {
            float re1[2][2], im1[2][2], re2[2][2], im2[2][2];
            pauli2(pi >> 2, re1, im1);
            pauli2(pi & 3, re2, im2);
            float Pre[4][4];
            for (int ra = 0; ra < 2; ++ra)
                for (int rb = 0; rb < 2; ++rb)
                    for (int ca = 0; ca < 2; ++ca)
                        for (int cb = 0; cb < 2; ++cb)
                            Pre[2 * ra + rb][2 * ca + cb] =
                                re1[ra][ca] * re2[rb][cb] - im1[ra][ca] * im2[rb][cb];
            for (int k = 0; k < 9; ++k) {
                int r = k >> 2, c = k & 3;
                sA[pi - 1][k] = 0.5f * (Pre[c][r] + Pre[r][c]);
            }
        }
    }
    __syncthreads();
    int o = threadIdx.x;
    if (o < COUT) {
        for (int c = 0; c < CIN; ++c)
            for (int j = 0; j < 9; ++j) {
                float s = (j == 0 || j == 5) ? 1.f : 0.f;   // t0[j]
                for (int i = 0; i < 15; ++i)
                    s += weight[(o * CIN + c) * 15 + i] * sA[i][j];
                weff[o * KP + c * 9 + j] = (_Float16)(0.25f * s);
            }
        for (int kk = CIN * 9; kk < KP; ++kk) weff[o * KP + kk] = (_Float16)0.f;
    }
}

// ---------------------------------------------------------------------------
// Main: implicit-GEMM 3x3 conv, async-LDS staging + V_WMMA_F32_16X16X32_F16.
// One block = one (b,h) row: 128 pixels x 64 out-channels. 8 wave32.
// ---------------------------------------------------------------------------
__global__ __launch_bounds__(256) void quanv_wmma_kernel(
        const float* __restrict__ x,
        const _Float16* __restrict__ weff,
        const float* __restrict__ bias,
        float* __restrict__ out) {
    extern __shared__ char smem[];

    const int tid  = threadIdx.x;
    const int wave = tid >> 5;
    const int lane = tid & 31;
    const int bh   = blockIdx.x;
    const int b    = bh >> 7;
    const int h    = bh & 127;

    const float* xb = x + (size_t)b * CIN * HW * HW;

    // --- async stage Weff (20480 B = 1280 x 16B, 5 per thread) ---
    #pragma unroll
    for (int i = 0; i < 5; ++i) {
        unsigned e = (unsigned)(i * 256 + tid) * 16u;
        async_copy_b128(LDS_W + e, (const char*)weff + e);
    }
    // --- async stage bias (256 B) ---
    if (tid < 16)
        async_copy_b128(LDS_BIAS + (unsigned)tid * 16u, (const char*)bias + tid * 16);

    // --- async stage raw input tile: 16 ch x 3 rows x 128 f32 (24 KB) ---
    // row rr = c*3 + r; wave handles rows [6*wave, 6*wave+6): uniform per wave.
    #pragma unroll
    for (int i = 0; i < 6; ++i) {
        int rr = wave * 6 + i;
        int c  = rr / 3;
        int r  = rr - c * 3;
        int hi = h + r - 1;
        if (hi >= 0 && hi < HW) {           // skip padded rows (zeroed at expand)
            unsigned off = LDS_RAW + (unsigned)rr * 512u + (unsigned)lane * 16u;
            async_copy_b128(off, xb + (size_t)c * HW * HW + (size_t)hi * HW + lane * 4);
        }
    }

    // warm L2 for the next output row's block while transfers are in flight
    if (h + 2 < HW)
        __builtin_prefetch(xb + (size_t)(tid & 15) * HW * HW +
                           (size_t)(h + 2) * HW + (tid >> 4) * 8, 0, 0);

    wait_async0();
    __syncthreads();

    // --- LDS->LDS im2col expansion: sB[n][kk] = raw[c][dy][n+dx-1] as f16 ---
    {
        const float* raw = (const float*)(smem + LDS_RAW);
        _Float16*    sB  = (_Float16*)(smem + LDS_B);
        #pragma unroll
        for (int e = tid; e < HW * KP; e += 256) {      // 80 iterations
            int kk = e >> 7;
            int n  = e & 127;
            float val = 0.f;
            if (kk < CIN * 9) {
                int c  = kk / 9;
                int j  = kk - c * 9;
                int dy = j / 3;
                int dx = j - dy * 3;
                int hi = h + dy - 1;
                int wi = n + dx - 1;
                if (hi >= 0 && hi < HW && wi >= 0 && wi < HW)
                    val = raw[(c * 3 + dy) * HW + wi];
            }
            sB[n * KP + kk] = (_Float16)val;
        }
    }
    __syncthreads();

    // --- WMMA: D[64 x 16] per wave, K = 5 x 32 ---
    const int hilo = lane >> 4;
    const int l16  = lane & 15;
    const int n0   = wave * 16;

    const _Float16* sW = (const _Float16*)smem;
    const _Float16* sB = (const _Float16*)(smem + LDS_B);

    v8f acc[4];
    #pragma unroll
    for (int mt = 0; mt < 4; ++mt)
        acc[mt] = (v8f){0.f, 0.f, 0.f, 0.f, 0.f, 0.f, 0.f, 0.f};

    const _Float16* bbase = &sB[(n0 + l16) * KP + hilo * 16];

    #pragma unroll
    for (int kt = 0; kt < 5; ++kt) {
        v16h bf = *(const v16h*)(bbase + kt * 32);      // 32B-aligned, contiguous K (k = 16*hilo + s), same as A
        #pragma unroll
        for (int mt = 0; mt < 4; ++mt) {
            // A layout: lanes 0-15 -> K{0..15}; lanes 16-31 -> K{16..31}
            const _Float16* abase = &sW[(mt * 16 + l16) * KP + kt * 32 + hilo * 16];
            v8h alo = *(const v8h*)(abase);
            v8h ahi = *(const v8h*)(abase + 8);
            v16h af = __builtin_shufflevector(alo, ahi,
                        0, 1, 2, 3, 4, 5, 6, 7, 8, 9, 10, 11, 12, 13, 14, 15);
            acc[mt] = __builtin_amdgcn_wmma_f32_16x16x32_f16(
                false, af, false, bf, (short)0, acc[mt], false, false);
        }
    }

    // --- epilogue: D lanes 0-15 -> M=v, lanes 16-31 -> M=v+8; add bias ---
    const float* sBias = (const float*)(smem + LDS_BIAS);
    float* ob = out + ((size_t)b * COUT) * (HW * HW) + (size_t)h * HW;
    #pragma unroll
    for (int mt = 0; mt < 4; ++mt) {
        #pragma unroll
        for (int v = 0; v < 8; ++v) {
            int o = mt * 16 + hilo * 8 + v;
            ob[(size_t)o * (HW * HW) + n0 + l16] = acc[mt][v] + sBias[o];
        }
    }
}

extern "C" void kernel_launch(void* const* d_in, const int* in_sizes, int n_in,
                              void* d_out, int out_size, void* d_ws, size_t ws_size,
                              hipStream_t stream) {
    const float* x      = (const float*)d_in[0];   // [B,16,128,128]
    const float* weight = (const float*)d_in[1];   // [64,16,15]
    const float* bias   = (const float*)d_in[2];   // [64,1]
    float* out          = (float*)d_out;           // [B,64,128,128]
    _Float16* weff      = (_Float16*)d_ws;         // [64][160] f16, 20 KB

    int B = in_sizes[0] / (CIN * HW * HW);

    hipLaunchKernelGGL(weff_setup_kernel, dim3(1), dim3(64), 0, stream,
                       weight, weff);
    hipLaunchKernelGGL(quanv_wmma_kernel, dim3(B * HW), dim3(256), LDS_TOTAL,
                       stream, x, weff, bias, out);
}
